// GeoFNO_4853313045055
// MI455X (gfx1250) — matrix operand order repaired and checked by reading, not compiled, not run on hardware
//
#include <hip/hip_runtime.h>
#include <hip/hip_bf16.h>
#include <math.h>

// GeoFNO forward for MI455X (gfx1250, wave32).
// Large GEMMs use v_wmma_f32_16x16x32_bf16 with 2x2 register blocking:
// one wave computes a 32x32 output block (4 accumulators), reusing each
// A/B fragment twice -> 16 flops/byte of fragment traffic.

constexpr int B_  = 4;
constexpr int N_  = 4096;   // gathered points
constexpr int NF_ = 8192;
constexpr int K_  = 512;    // Fourier modes
constexpr int C_  = 128;    // channels
constexpr int L_  = 4;      // layers
constexpr int IN_ = 3;
constexpr int FC_ = 128;

typedef __attribute__((ext_vector_type(16))) __bf16 bf16x16;
typedef __attribute__((ext_vector_type(8)))  __bf16 bf16x8;
typedef __attribute__((ext_vector_type(8)))  float  f32x8;

#define DI static __device__ __forceinline__

DI float gelu_exact(float v) {
  return 0.5f * v * (1.0f + erff(v * 0.70710678118654752440f));
}

// ---- WMMA fragment helpers (ISA 7.12.2 layouts, wave32) -------------------

// A-matrix 16x32 bf16, row-major, lda in elements.
// lanes 0-15: row=lane, K chunks [0..7] and [16..23]
// lanes 16-31: row=lane-16, K chunks [8..15] and [24..31]
DI bf16x16 load_a16x32(const __bf16* A, int lda, int lane) {
  const int row = lane & 15;
  const int sel = (lane >> 4) & 1;
  const __bf16* p = A + (size_t)row * lda + sel * 8;
  bf16x8 lo = *(const bf16x8*)(p);
  bf16x8 hi = *(const bf16x8*)(p + 16);
  return __builtin_shufflevector(lo, hi, 0,1,2,3,4,5,6,7,8,9,10,11,12,13,14,15);
}

// B-matrix 32x16 bf16 stored column-contiguous: B[kk,col] = Bp[col*ldb + kk].
// lanes 0-15: col=lane, K [0..7] and [16..23] ; lanes 16-31: col=lane-16, K [8..15] and [24..31]
DI bf16x16 load_b32x16(const __bf16* Bp, int ldb, int lane) {
  const int col  = lane & 15;
  const int half = (lane >> 4) & 1;
  const __bf16* p = Bp + (size_t)col * ldb + half * 8;
  bf16x8 lo = *(const bf16x8*)(p);
  bf16x8 hi = *(const bf16x8*)(p + 16);
  return __builtin_shufflevector(lo, hi, 0,1,2,3,4,5,6,7,8,9,10,11,12,13,14,15);
}

// C/D 16x16 f32: VGPR r -> (M = r + 8*(lane>=16), N = lane&15)
DI void store_c16x16(float* Cp, int ldc, int lane, f32x8 acc) {
  const int col  = lane & 15;
  const int rofs = ((lane >> 4) & 1) * 8;
#pragma unroll
  for (int r = 0; r < 8; ++r) Cp[(size_t)(r + rofs) * ldc + col] = acc[r];
}

DI f32x8 wmma_bf16(bf16x16 a, bf16x16 b, f32x8 c) {
  return __builtin_amdgcn_wmma_f32_16x16x32_bf16(false, a, false, b, (short)0, c,
                                                 false, false);
}

// ---- Precompute kernels ---------------------------------------------------

// Bases at the gathered columns ind[n]:
//   wbC/wbS  : [b][k][n]  cos/sin * (wq*N)   (forward-transform B operand)
//   bCt/bSt  : [b][n][k]  cos/sin            (inverse-transform B operand)
__global__ void k_bases(const float* __restrict__ xf, const int* __restrict__ ind,
                        const float* __restrict__ modes,
                        __hip_bfloat16* wbC, __hip_bfloat16* wbS,
                        __hip_bfloat16* bCt, __hip_bfloat16* bSt) {
  size_t t = (size_t)blockIdx.x * blockDim.x + threadIdx.x;
  if (t >= (size_t)B_ * K_ * N_) return;
  int n = (int)(t % N_);
  int k = (int)((t / N_) % K_);
  int b = (int)(t / ((size_t)N_ * K_));
  const float* xp = xf + ((size_t)b * NF_ + (size_t)ind[n]) * 4;
  float g0 = xp[0], g1 = xp[1], wq = xp[2];
  float th = modes[2 * k] * g0 + modes[2 * k + 1] * g1;
  float sv, cv;
  __sincosf(th, &sv, &cv);
  float sc = wq * (float)N_;
  wbC[t] = __float2bfloat16(cv * sc);          // t == (b*K+k)*N + n
  wbS[t] = __float2bfloat16(sv * sc);
  size_t ti = ((size_t)b * N_ + n) * K_ + k;
  bCt[ti] = __float2bfloat16(cv);
  bSt[ti] = __float2bfloat16(sv);
}

__global__ void k_cvt_bf16(const float* __restrict__ w, __hip_bfloat16* wb, int count) {
  int t = blockIdx.x * blockDim.x + threadIdx.x;
  if (t < count) wb[t] = __float2bfloat16(w[t]);
}

// fc0: h[b][c][n] = sum_i x[b][n][i] * fc0_w[i][c] + fc0_b[c]
__global__ void k_fc0(const float* __restrict__ x, const float* __restrict__ w,
                      const float* __restrict__ bias,
                      float* h, __hip_bfloat16* hbf, __hip_bfloat16* hT) {
  size_t t = (size_t)blockIdx.x * blockDim.x + threadIdx.x;
  if (t >= (size_t)B_ * C_ * N_) return;
  int n = (int)(t % N_);
  int c = (int)((t / N_) % C_);
  int b = (int)(t / ((size_t)N_ * C_));
  const float* xp = x + ((size_t)b * N_ + n) * IN_;
  float v = bias[c];
#pragma unroll
  for (int i = 0; i < IN_; ++i) v += xp[i] * w[i * C_ + c];
  h[t] = v;
  __hip_bfloat16 bv = __float2bfloat16(v);
  hbf[t] = bv;
  hT[((size_t)b * N_ + n) * C_ + c] = bv;
}

// ---- Per-layer WMMA GEMMs (2x2 register-blocked) --------------------------

// Forward transform: Xc/S[b][c][k] = sum_n hbf[b][c][n] * wb{C,S}[b][k][n]
// One wave -> 32x32 block of (c,k). grid(4*16, B, 2{cos,sin}), block 32.
__global__ void __launch_bounds__(32)
k_fwd_gemm(const __hip_bfloat16* __restrict__ hbf,
           const __hip_bfloat16* __restrict__ wbC,
           const __hip_bfloat16* __restrict__ wbS,
           float* __restrict__ Xc, float* __restrict__ S) {
  const int lane = threadIdx.x;
  const int nb = K_ / 32;
  const int mb = blockIdx.x / nb;
  const int kb = blockIdx.x % nb;
  const int b  = blockIdx.y;
  const bool sn = (blockIdx.z != 0);
  const __bf16* A0 = (const __bf16*)hbf + ((size_t)b * C_ + mb * 32) * N_;
  const __bf16* A1 = A0 + (size_t)16 * N_;
  const __bf16* B0 = (const __bf16*)(sn ? wbS : wbC) + ((size_t)b * K_ + kb * 32) * N_;
  const __bf16* B1 = B0 + (size_t)16 * N_;
  f32x8 c00 = {}, c01 = {}, c10 = {}, c11 = {};
  for (int ks = 0; ks < N_; ks += 32) {
    __builtin_prefetch(A0 + ks + 512, 0, 3);
    __builtin_prefetch(B0 + ks + 512, 0, 3);
    bf16x16 a0 = load_a16x32(A0 + ks, N_, lane);
    bf16x16 a1 = load_a16x32(A1 + ks, N_, lane);
    bf16x16 b0 = load_b32x16(B0 + ks, N_, lane);
    bf16x16 b1 = load_b32x16(B1 + ks, N_, lane);
    c00 = wmma_bf16(a0, b0, c00);
    c01 = wmma_bf16(a0, b1, c01);
    c10 = wmma_bf16(a1, b0, c10);
    c11 = wmma_bf16(a1, b1, c11);
  }
  float* O = (sn ? S : Xc) + ((size_t)b * C_ + mb * 32) * K_ + kb * 32;
  store_c16x16(O,                     K_, lane, c00);
  store_c16x16(O + 16,                K_, lane, c01);
  store_c16x16(O + (size_t)16 * K_,      K_, lane, c10);
  store_c16x16(O + (size_t)16 * K_ + 16, K_, lane, c11);
}

// Zero-mode forward: x0h[b][c] = sum_n h[b][c][n] * (wq[b,ind[n]] * N)
__global__ void k_x0h(const float* __restrict__ h, const float* __restrict__ xf,
                      const int* __restrict__ ind, float* __restrict__ x0h) {
  __shared__ float red[256];
  const int bc = blockIdx.x;
  const int b = bc / C_;
  const float* hp  = h + (size_t)bc * N_;
  const float* xfb = xf + (size_t)b * NF_ * 4;
  float acc = 0.f;
  for (int n = threadIdx.x; n < N_; n += 256)
    acc += hp[n] * (xfb[(size_t)ind[n] * 4 + 2] * (float)N_);
  red[threadIdx.x] = acc;
  __syncthreads();
  for (int s = 128; s > 0; s >>= 1) {
    if (threadIdx.x < s) red[threadIdx.x] += red[threadIdx.x + s];
    __syncthreads();
  }
  if (threadIdx.x == 0) x0h[bc] = red[0];
}

// Mode mixing. With S = +sum(xg*wbS) (so xsh = -S):
//   Fc = 2*fch = (2/NF) * sum_i (Xc*wc + S*ws)
//   Fs = -2*fsh = (2/NF) * sum_i (S*wc - Xc*ws)
// so that x1 = f0h + Fc.basC + Fs.basS.
__global__ void k_mix(const float* __restrict__ Xc, const float* __restrict__ S,
                      const float* __restrict__ wc, const float* __restrict__ ws,
                      __hip_bfloat16* Fc, __hip_bfloat16* Fs, int layer) {
  size_t t = (size_t)blockIdx.x * blockDim.x + threadIdx.x;
  if (t >= (size_t)B_ * C_ * K_) return;
  int k = (int)(t % K_);
  int o = (int)((t / K_) % C_);
  int b = (int)(t / ((size_t)K_ * C_));
  const float* wcl = wc + (size_t)layer * C_ * C_ * K_ + (size_t)o * K_ + k;
  const float* wsl = ws + (size_t)layer * C_ * C_ * K_ + (size_t)o * K_ + k;
  const float* Xb  = Xc + (size_t)b * C_ * K_ + k;
  const float* Sb  = S  + (size_t)b * C_ * K_ + k;
  float aA = 0.f, aB = 0.f;
  for (int i = 0; i < C_; ++i) {
    float wci = wcl[(size_t)i * C_ * K_];
    float wsi = wsl[(size_t)i * C_ * K_];
    float xc  = Xb[(size_t)i * K_];
    float s   = Sb[(size_t)i * K_];
    aA += xc * wci + s * wsi;
    aB += s * wci - xc * wsi;
  }
  const float sc = 2.0f / (float)NF_;
  Fc[t] = __float2bfloat16(aA * sc);   // t == (b*C+o)*K + k
  Fs[t] = __float2bfloat16(aB * sc);
}

__global__ void k_f0h(const float* __restrict__ x0h, const float* __restrict__ w0,
                      float* __restrict__ f0h, int layer) {
  int t = blockIdx.x * blockDim.x + threadIdx.x;
  if (t >= B_ * C_) return;
  int o = t % C_, b = t / C_;
  const float* w = w0 + (size_t)layer * C_ * C_;
  float a = 0.f;
  for (int i = 0; i < C_; ++i) a += x0h[b * C_ + i] * w[i * C_ + o];
  f0h[t] = a / (float)NF_;
}

// Inverse transform: y[b][o][n] = Fc.basC + Fs.basS
// One wave -> 32x32 block of (o,n). grid(4*128, B), block 32.
__global__ void __launch_bounds__(32)
k_inv_gemm(const __hip_bfloat16* __restrict__ Fc, const __hip_bfloat16* __restrict__ Fs,
           const __hip_bfloat16* __restrict__ bCt, const __hip_bfloat16* __restrict__ bSt,
           float* __restrict__ y) {
  const int lane = threadIdx.x;
  const int nb = N_ / 32;
  const int mb = blockIdx.x / nb;
  const int nt = blockIdx.x % nb;
  const int b  = blockIdx.y;
  const __bf16* Ac0 = (const __bf16*)Fc  + ((size_t)b * C_ + mb * 32) * K_;
  const __bf16* Ac1 = Ac0 + (size_t)16 * K_;
  const __bf16* As0 = (const __bf16*)Fs  + ((size_t)b * C_ + mb * 32) * K_;
  const __bf16* As1 = As0 + (size_t)16 * K_;
  const __bf16* Bc0 = (const __bf16*)bCt + ((size_t)b * N_ + nt * 32) * K_;
  const __bf16* Bc1 = Bc0 + (size_t)16 * K_;
  const __bf16* Bs0 = (const __bf16*)bSt + ((size_t)b * N_ + nt * 32) * K_;
  const __bf16* Bs1 = Bs0 + (size_t)16 * K_;
  f32x8 c00 = {}, c01 = {}, c10 = {}, c11 = {};
  for (int kk = 0; kk < K_; kk += 32) {
    bf16x16 a0 = load_a16x32(Ac0 + kk, K_, lane);
    bf16x16 a1 = load_a16x32(Ac1 + kk, K_, lane);
    bf16x16 b0 = load_b32x16(Bc0 + kk, K_, lane);
    bf16x16 b1 = load_b32x16(Bc1 + kk, K_, lane);
    c00 = wmma_bf16(a0, b0, c00);
    c01 = wmma_bf16(a0, b1, c01);
    c10 = wmma_bf16(a1, b0, c10);
    c11 = wmma_bf16(a1, b1, c11);
    a0 = load_a16x32(As0 + kk, K_, lane);
    a1 = load_a16x32(As1 + kk, K_, lane);
    b0 = load_b32x16(Bs0 + kk, K_, lane);
    b1 = load_b32x16(Bs1 + kk, K_, lane);
    c00 = wmma_bf16(a0, b0, c00);
    c01 = wmma_bf16(a0, b1, c01);
    c10 = wmma_bf16(a1, b0, c10);
    c11 = wmma_bf16(a1, b1, c11);
  }
  float* O = y + ((size_t)b * C_ + mb * 32) * N_ + nt * 32;
  store_c16x16(O,                     N_, lane, c00);
  store_c16x16(O + 16,                N_, lane, c01);
  store_c16x16(O + (size_t)16 * N_,      N_, lane, c10);
  store_c16x16(O + (size_t)16 * N_ + 16, N_, lane, c11);
}

// Channel conv: z[b][o][n] = sum_i conv_w[l][o][i] * h[b][i][n]  (hT is [n][c])
// One wave -> 32x32 block. grid(4*128, B), block 32.
__global__ void __launch_bounds__(32)
k_conv_gemm(const __hip_bfloat16* __restrict__ cw, const __hip_bfloat16* __restrict__ hT,
            float* __restrict__ z, int layer) {
  const int lane = threadIdx.x;
  const int nb = N_ / 32;
  const int mb = blockIdx.x / nb;
  const int nt = blockIdx.x % nb;
  const int b  = blockIdx.y;
  const __bf16* A0 = (const __bf16*)cw + (size_t)layer * C_ * C_ + (size_t)(mb * 32) * C_;
  const __bf16* A1 = A0 + (size_t)16 * C_;
  const __bf16* B0 = (const __bf16*)hT + ((size_t)b * N_ + nt * 32) * C_;
  const __bf16* B1 = B0 + (size_t)16 * C_;
  f32x8 c00 = {}, c01 = {}, c10 = {}, c11 = {};
#pragma unroll
  for (int kk = 0; kk < C_; kk += 32) {
    bf16x16 a0 = load_a16x32(A0 + kk, C_, lane);
    bf16x16 a1 = load_a16x32(A1 + kk, C_, lane);
    bf16x16 b0 = load_b32x16(B0 + kk, C_, lane);
    bf16x16 b1 = load_b32x16(B1 + kk, C_, lane);
    c00 = wmma_bf16(a0, b0, c00);
    c01 = wmma_bf16(a0, b1, c01);
    c10 = wmma_bf16(a1, b0, c10);
    c11 = wmma_bf16(a1, b1, c11);
  }
  float* O = z + ((size_t)b * C_ + mb * 32) * N_ + nt * 32;
  store_c16x16(O,                     N_, lane, c00);
  store_c16x16(O + 16,                N_, lane, c01);
  store_c16x16(O + (size_t)16 * N_,      N_, lane, c10);
  store_c16x16(O + (size_t)16 * N_ + 16, N_, lane, c11);
}

// h_new = [gelu]( y + z + f0h[b,c] + conv_b[l,c] ); also refresh bf16 copies.
__global__ void k_combine(const float* __restrict__ y, const float* __restrict__ z,
                          const float* __restrict__ f0h, const float* __restrict__ convb,
                          float* h, __hip_bfloat16* hbf, __hip_bfloat16* hT,
                          int layer, int last) {
  size_t t = (size_t)blockIdx.x * blockDim.x + threadIdx.x;
  if (t >= (size_t)B_ * C_ * N_) return;
  int n = (int)(t % N_);
  int c = (int)((t / N_) % C_);
  int b = (int)(t / ((size_t)N_ * C_));
  float v = y[t] + z[t] + f0h[b * C_ + c] + convb[layer * C_ + c];
  if (!last) v = gelu_exact(v);
  h[t] = v;
  __hip_bfloat16 bv = __float2bfloat16(v);
  hbf[t] = bv;
  hT[((size_t)b * N_ + n) * C_ + c] = bv;
}

// Final MLP: out[b,n] = gelu(h[:,n] @ fc1_w + fc1_b) @ fc2_w + fc2_b
__global__ void k_fc_final(const float* __restrict__ h, const float* __restrict__ fc1w,
                           const float* __restrict__ fc1b, const float* __restrict__ fc2w,
                           const float* __restrict__ fc2b, float* __restrict__ out) {
  __shared__ float red[FC_];
  const int bn = blockIdx.x;
  const int b = bn / N_, n = bn % N_;
  const int f = threadIdx.x;
  const float* hp = h + (size_t)b * C_ * N_ + n;
  float v = fc1b[f];
  for (int c = 0; c < C_; ++c) v += hp[(size_t)c * N_] * fc1w[c * FC_ + f];
  v = gelu_exact(v) * fc2w[f];
  red[f] = v;
  __syncthreads();
  for (int s = FC_ / 2; s > 0; s >>= 1) {
    if (f < s) red[f] += red[f + s];
    __syncthreads();
  }
  if (f == 0) out[bn] = red[0] + fc2b[0];
}

// ---- Host launcher --------------------------------------------------------

extern "C" void kernel_launch(void* const* d_in, const int* in_sizes, int n_in,
                              void* d_out, int out_size, void* d_ws, size_t ws_size,
                              hipStream_t stream) {
  const float* x      = (const float*)d_in[0];
  const float* xf     = (const float*)d_in[1];
  const int*   ind    = (const int*)  d_in[2];
  const float* modes  = (const float*)d_in[3];
  const float* fc0_w  = (const float*)d_in[4];
  const float* fc0_b  = (const float*)d_in[5];
  const float* wc     = (const float*)d_in[6];
  const float* ws_    = (const float*)d_in[7];
  const float* w0     = (const float*)d_in[8];
  const float* conv_w = (const float*)d_in[9];
  const float* conv_b = (const float*)d_in[10];
  const float* fc1_w  = (const float*)d_in[11];
  const float* fc1_b  = (const float*)d_in[12];
  const float* fc2_w  = (const float*)d_in[13];
  const float* fc2_b  = (const float*)d_in[14];
  float* out = (float*)d_out;

  char* base = (char*)d_ws;
  size_t off = 0;
  auto take = [&](size_t bytes) -> char* {
    char* p = base + off;
    off += (bytes + 255) & ~(size_t)255;
    return p;
  };
  const size_t BKN = (size_t)B_ * K_ * N_;
  const size_t BCN = (size_t)B_ * C_ * N_;
  const size_t BCK = (size_t)B_ * C_ * K_;

  __hip_bfloat16* wbC  = (__hip_bfloat16*)take(BKN * 2);
  __hip_bfloat16* wbS  = (__hip_bfloat16*)take(BKN * 2);
  __hip_bfloat16* bCt  = (__hip_bfloat16*)take(BKN * 2);
  __hip_bfloat16* bSt  = (__hip_bfloat16*)take(BKN * 2);
  float*          h    = (float*)take(BCN * 4);
  __hip_bfloat16* hbf  = (__hip_bfloat16*)take(BCN * 2);
  __hip_bfloat16* hT   = (__hip_bfloat16*)take(BCN * 2);
  float*          Xc   = (float*)take(BCK * 4);
  float*          S    = (float*)take(BCK * 4);
  __hip_bfloat16* Fc   = (__hip_bfloat16*)take(BCK * 2);
  __hip_bfloat16* Fs   = (__hip_bfloat16*)take(BCK * 2);
  float*          x0h  = (float*)take((size_t)B_ * C_ * 4);
  float*          f0h  = (float*)take((size_t)B_ * C_ * 4);
  float*          y    = (float*)take(BCN * 4);
  float*          z    = (float*)take(BCN * 4);
  __hip_bfloat16* cwbf = (__hip_bfloat16*)take((size_t)L_ * C_ * C_ * 2);
  (void)ws_size; (void)in_sizes; (void)n_in; (void)out_size;

  k_cvt_bf16<<<(L_ * C_ * C_ + 255) / 256, 256, 0, stream>>>(conv_w, cwbf, L_ * C_ * C_);
  k_bases<<<(unsigned)((BKN + 255) / 256), 256, 0, stream>>>(xf, ind, modes, wbC, wbS, bCt, bSt);
  k_fc0<<<(unsigned)((BCN + 255) / 256), 256, 0, stream>>>(x, fc0_w, fc0_b, h, hbf, hT);

  for (int l = 0; l < L_; ++l) {
    k_fwd_gemm<<<dim3((C_ / 32) * (K_ / 32), B_, 2), 32, 0, stream>>>(hbf, wbC, wbS, Xc, S);
    k_x0h<<<B_ * C_, 256, 0, stream>>>(h, xf, ind, x0h);
    k_mix<<<(unsigned)((BCK + 255) / 256), 256, 0, stream>>>(Xc, S, wc, ws_, Fc, Fs, l);
    k_f0h<<<(B_ * C_ + 255) / 256, 256, 0, stream>>>(x0h, w0, f0h, l);
    k_inv_gemm<<<dim3((C_ / 32) * (N_ / 32), B_), 32, 0, stream>>>(Fc, Fs, bCt, bSt, y);
    k_conv_gemm<<<dim3((C_ / 32) * (N_ / 32), B_), 32, 0, stream>>>(cwbf, hT, z, l);
    k_combine<<<(unsigned)((BCN + 255) / 256), 256, 0, stream>>>(y, z, f0h, conv_b, h, hbf, hT,
                                                                 l, (l == L_ - 1) ? 1 : 0);
  }
  k_fc_final<<<B_ * N_, FC_, 0, stream>>>(h, fc1_w, fc1_b, fc2_w, fc2_b, out);
}
